// GLVQ_49520972923161
// MI455X (gfx1250) — matrix operand order repaired and checked by reading, not compiled, not run on hardware
//
#include <hip/hip_runtime.h>
#include <stdint.h>

#define N_TOTAL   262144
#define DIM       64
#define NCLASS    128
#define NPROTO    512
#define WG        256
#define NWAVE     8
#define GRID_MAIN 512
#define PASSES    4            // N_TOTAL / (GRID_MAIN * 128)

typedef __attribute__((ext_vector_type(16))) __bf16 v16bf;
typedef __attribute__((ext_vector_type(8)))  float  v8f;

#define BIG 3.0e38f            // finite "infinity" sentinel (real dists are O(100))

__device__ __forceinline__ unsigned short f2bf(float f) {
  union { float f; uint32_t u; } c{f};
  uint32_t u = c.u;
  u += 0x7FFFu + ((u >> 16) & 1u);          // round-to-nearest-even
  return (unsigned short)(u >> 16);
}

// min with fast-math flags; inputs canonical (arith results) -> bare v_min_num_f32
__device__ __forceinline__ float fmin_nc(float a, float b) {
#pragma float_control(precise, off)
  return __builtin_fminf(a, b);
}

// forced single-instruction min (for shuffle-sourced values the backend
// cannot prove canonical and would otherwise canonicalize)
__device__ __forceinline__ float vmin_asm(float a, float b) {
  float d;
  asm("v_min_num_f32 %0, %1, %2" : "=v"(d) : "v"(a), "v"(b));
  return d;
}

// ---- prep: prototypes fp32 -> bf16 of (-2*p) (row major) + ||p||^2, into ws ----
__global__ void glvq_prep(const float* __restrict__ protos,
                          unsigned short* __restrict__ pbf,
                          float* __restrict__ psq) {
  int row = blockIdx.x;              // 512 rows
  int t   = threadIdx.x;             // 32 threads
  float f0 = protos[row * DIM + t];
  float f1 = protos[row * DIM + 32 + t];
  pbf[row * DIM + t]      = f2bf(-2.0f * f0);   // fold the -2 of the distance here
  pbf[row * DIM + 32 + t] = f2bf(-2.0f * f1);
  float s = f0 * f0 + f1 * f1;
#pragma unroll
  for (int o = 16; o > 0; o >>= 1) s += __shfl_xor(s, o, 32);
  if (t == 0) psq[row] = s;
}

// ---- main: per 16-row tile per wave, WMMA vs all 512 protos, dual-min ----
__launch_bounds__(WG)
__global__ void glvq_main(const float* __restrict__ x,
                          const int* __restrict__ y,
                          const unsigned short* __restrict__ ws_base, // pbf, then psq
                          float* __restrict__ partial) {
  __shared__ __align__(16) unsigned char smem[NPROTO * DIM * 2 + NPROTO * 4]; // 66 KB
  __shared__ float wsum[NWAVE];
  unsigned short* pbf = (unsigned short*)smem;
  const float*    psq = (const float*)(smem + NPROTO * DIM * 2);

  const int tid  = threadIdx.x;
  const int wave = tid >> 5;
  const int lane = tid & 31;
  const int h    = lane >> 4;        // half: selects K-offset per ISA layouts
  const int lm   = lane & 15;        // column / row-within-tile index

  // ---- async stage 66KB (bf16 -2p + psq) global -> LDS ----
  {
    const int nvec = (NPROTO * DIM * 2 + NPROTO * 4) / 16;   // 4224 x 16B
    uint64_t gbase = (uint64_t)(uintptr_t)ws_base;
    uint32_t lbase = (uint32_t)(uintptr_t)smem;
    for (int i = tid; i < nvec; i += WG) {
      uint32_t voff  = (uint32_t)i * 16u;
      uint32_t laddr = lbase + voff;
      asm volatile("global_load_async_to_lds_b128 %0, %1, %2"
                   :: "v"(laddr), "v"(voff), "s"(gbase) : "memory");
    }
    asm volatile("s_wait_asynccnt 0" ::: "memory");
  }
  __syncthreads();

  float acc = 0.0f;

#pragma unroll 1
  for (int pass = 0; pass < PASSES; ++pass) {
    const int rowBase = blockIdx.x * (PASSES * 128) + pass * 128 + wave * 16;
    const float4* x4 = (const float4*)(x + (size_t)(rowBase + lm) * DIM);

    // lane's 32 fp32 elements of row (rowBase+lm); partner lane (^16) has the rest
    float4 t0 = x4[2*h],      t1 = x4[2*h + 1];
    float4 t2 = x4[4 + 2*h],  t3 = x4[4 + 2*h + 1];
    float4 t4 = x4[8 + 2*h],  t5 = x4[8 + 2*h + 1];
    float4 t6 = x4[12 + 2*h], t7 = x4[12 + 2*h + 1];

    float sq = t0.x*t0.x + t0.y*t0.y + t0.z*t0.z + t0.w*t0.w
             + t1.x*t1.x + t1.y*t1.y + t1.z*t1.z + t1.w*t1.w
             + t2.x*t2.x + t2.y*t2.y + t2.z*t2.z + t2.w*t2.w
             + t3.x*t3.x + t3.y*t3.y + t3.z*t3.z + t3.w*t3.w
             + t4.x*t4.x + t4.y*t4.y + t4.z*t4.z + t4.w*t4.w
             + t5.x*t5.x + t5.y*t5.y + t5.z*t5.z + t5.w*t5.w
             + t6.x*t6.x + t6.y*t6.y + t6.z*t6.z + t6.w*t6.w
             + t7.x*t7.x + t7.y*t7.y + t7.z*t7.z + t7.w*t7.w;
    sq += __shfl_xor(sq, 16, 32);    // full ||x_row||^2

    // A = x in bf16, per ISA 16-bit A 16x32 layout (chunk0: K 0..31, chunk1: K 32..63)
    union { v16bf v; unsigned short s[16]; } A0, A1;
#define PK4(U, b, q) U.s[(b)+0]=f2bf((q).x); U.s[(b)+1]=f2bf((q).y); \
                     U.s[(b)+2]=f2bf((q).z); U.s[(b)+3]=f2bf((q).w);
    PK4(A0, 0, t0) PK4(A0, 4, t1) PK4(A0, 8, t2) PK4(A0, 12, t3)
    PK4(A1, 0, t4) PK4(A1, 4, t5) PK4(A1, 8, t6) PK4(A1, 12, t7)
#undef PK4

    // per-row class-hit metadata for this lane's 8 C-rows (M = 8h+i)
    int yv = y[rowBase + lm];
    int shit_i[8];
#pragma unroll
    for (int i = 0; i < 8; ++i) {
      int yi = __shfl(yv, 8*h + i, 32);
      // all 4 same-class protos of class yi live in lane lm=yi&15, group s=(yi>>4)&7
      shit_i[i] = ((yi & 15) == lm) ? ((yi >> 4) & 7) : -1;
    }

    float d1e[8], d2e[8];
#pragma unroll
    for (int i = 0; i < 8; ++i) { d1e[i] = BIG; d2e[i] = BIG; }

    const uint4* bmat = (const uint4*)pbf;
    // two class-groups in flight: fills WMMA->VALU hazard slots without spilling
#pragma unroll 2
    for (int s = 0; s < 8; ++s) {          // class-group
      float g[8];
#pragma unroll
      for (int j = 0; j < 4; ++j) {        // 4 protos of this class on this lane
        const int t = s + 8 * j;
        const int n = lm + 16 * t;         // proto column
        union { uint4 q[2]; v16bf v; } B0, B1;   // dense 32x16 bf16 B layout
        B0.q[0] = bmat[n * 8 + h];         B0.q[1] = bmat[n * 8 + 2 + h];
        B1.q[0] = bmat[n * 8 + 4 + h];     B1.q[1] = bmat[n * 8 + 6 + h];
        v8f c8 = {};                       // inline-0 accumulator (src2 = 0)
        c8 = __builtin_amdgcn_wmma_f32_16x16x32_bf16(false, A0.v, false, B0.v,
                                                     (short)0, c8, false, false);
        c8 = __builtin_amdgcn_wmma_f32_16x16x32_bf16(false, A1.v, false, B1.v,
                                                     (short)0, c8, false, false);
        float pq = psq[n];                 // ||p||^2 added post-WMMA (canonical adds)
#pragma unroll
        for (int i = 0; i < 8; ++i) {
          float e = c8[i] + pq;            // e = ||p||^2 - 2 x.p
          g[i] = (j == 0) ? e : fmin_nc(g[i], e);
        }
      }
#pragma unroll
      for (int i = 0; i < 8; ++i) {
        bool hit = (s == shit_i[i]);
        d1e[i] = hit ? g[i] : d1e[i];               // class-pure group: select, no min
        d2e[i] = fmin_nc(d2e[i], hit ? BIG : g[i]);
      }
    }

    // min-reduce across the 16 lanes of each half (each lane = 32 protos)
#pragma unroll
    for (int o = 1; o < 16; o <<= 1) {
#pragma unroll
      for (int i = 0; i < 8; ++i) {
        d1e[i] = vmin_asm(d1e[i], __shfl_xor(d1e[i], o, 32));
        d2e[i] = vmin_asm(d2e[i], __shfl_xor(d2e[i], o, 32));
      }
    }

    if (lm == 0) {
#pragma unroll
      for (int i = 0; i < 8; ++i) {
        float xsq = __shfl(sq, 8*h + i, 32);   // ||x||^2 shift applied post-min
        float d1 = xsq + d1e[i];
        float d2 = xsq + d2e[i];
        float mu = (d1 - d2) / (d1 + d2 + 1e-9f);
        acc += 1.0f / (1.0f + __expf(-mu));
      }
    }
  }

  acc += __shfl_xor(acc, 16, 32);    // only lanes 0,16 were nonzero
  if (lane == 0) wsum[wave] = acc;
  __syncthreads();
  if (tid == 0) {
    float s = 0.0f;
#pragma unroll
    for (int w = 0; w < NWAVE; ++w) s += wsum[w];
    partial[blockIdx.x] = s;
  }
}

// ---- deterministic final reduction ----
__global__ void glvq_finish(const float* __restrict__ partial, float* __restrict__ out) {
  int t = threadIdx.x;               // 32 threads
  float s = 0.0f;
  for (int k = t; k < GRID_MAIN; k += 32) s += partial[k];
#pragma unroll
  for (int o = 16; o > 0; o >>= 1) s += __shfl_xor(s, o, 32);
  if (t == 0) out[0] = s * (1.0f / (float)N_TOTAL);
}

extern "C" void kernel_launch(void* const* d_in, const int* in_sizes, int n_in,
                              void* d_out, int out_size, void* d_ws, size_t ws_size,
                              hipStream_t stream) {
  const float* x      = (const float*)d_in[0];
  const float* protos = (const float*)d_in[1];
  const int*   y      = (const int*)d_in[2];
  float* out = (float*)d_out;

  unsigned short* pbf = (unsigned short*)d_ws;
  float* psq     = (float*)((char*)d_ws + NPROTO * DIM * 2);
  float* partial = (float*)((char*)d_ws + NPROTO * DIM * 2 + NPROTO * 4);

  glvq_prep<<<NPROTO, 32, 0, stream>>>(protos, pbf, psq);
  glvq_main<<<GRID_MAIN, WG, 0, stream>>>(x, y, (const unsigned short*)d_ws, partial);
  glvq_finish<<<1, 32, 0, stream>>>(partial, out);
}
